// Classifier_68556267978945
// MI455X (gfx1250) — matrix operand order repaired and checked by reading, not compiled, not run on hardware
//
#include <hip/hip_runtime.h>
#include <hip/hip_bf16.h>
#include <math.h>

// ---------------------------------------------------------------------------
// ViT-ish classifier forward for MI455X (gfx1250, wave32, WMMA + TDM).
// One workgroup (256 threads = 8 waves) per batch sample; all activations in
// LDS; every GEMM via v_wmma_f32_16x16x32_f16 (f16 operands, f32 accum).
// Phase-1 input chunks are DMA'd global->LDS with tensor_load_to_lds (double
// buffered, overlapped with WMMA); all LDS fragment traffic is 128-bit.
// ---------------------------------------------------------------------------

typedef __attribute__((ext_vector_type(16))) _Float16 v16h;
typedef __attribute__((ext_vector_type(8)))  _Float16 v8h;
typedef __attribute__((ext_vector_type(8)))  float    v8f;

#define NMSA   3
#define LN_EPS 1e-5f
#define ZERO8 ((v8f){0.f,0.f,0.f,0.f,0.f,0.f,0.f,0.f})

// ---- LDS layout (bytes), total 275456 (< 320 KB/WGP) ----------------------
#define OFF_X   0        // float [256][64]            65536
#define OFF_XN  65536    // half  [256][64]            32768  (also A-chunk)
#define OFF_Q   98304    // half  [256][32]            16384
#define OFF_K   114688   // half  [256][32]            16384
#define OFF_V   131072   // half  Vt [32][256]         16384  (transposed)
#define OFF_S   147456   // half  [128][256]           65536  (also T1/T2)
#define OFF_H   212992   // half  [256][96]            49152
#define OFF_W   262144   // half  staging (<=6144)     12288  (transposed [N][K])
#define OFF_R   274432   // float [256]                 1024
// Phase-1 only: two 64KB raw-x staging buffers overlay the attention region.
#define OFF_STG 98304    // float [2][16384]          131072 (ends 229376)
#define SMEM_BYTES 275456

__device__ __forceinline__ v8f wmma16(v16h a, v16h b, v8f c) {
  return __builtin_amdgcn_wmma_f32_16x16x32_f16(
      /*neg_a=*/false, a, /*neg_b=*/false, b,
      /*c_mod=*/(short)0, c, /*reuse_a=*/false, /*reuse_b=*/false);
}

__device__ __forceinline__ v16h cat8(v8h lo, v8h hi) {
  return __builtin_shufflevector(lo, hi, 0, 1, 2, 3, 4, 5, 6, 7,
                                 8, 9, 10, 11, 12, 13, 14, 15);
}

// A fragment, 16-bit 16x32 (MxK), source row-major [M][lda].
// v16h elements 0..7 = K base..base+7, 8..15 = K base+16..base+23
// => two 16-byte ds loads. (lda, kBase multiples of 16 halves)
__device__ __forceinline__ v16h load_a_frag(const _Float16* A, int lda,
                                            int mBase, int kBase, int lane) {
  const _Float16* p =
      A + (mBase + (lane & 15)) * lda + kBase + (lane >> 4) * 8;
  v8h lo = *(const v8h*)p;         // K = base .. base+7
  v8h hi = *(const v8h*)(p + 16);  // K = base+16 .. base+23
  return cat8(lo, hi);
}

// B fragment, 16-bit 32x16 (KxN), with source stored TRANSPOSED row-major
// [N][ldk] (i.e. src[n*ldk + k] == B[k][n]). Lane needs K = hs*8 + 0..7 and
// 16 + hs*8 + 0..7 for its column n -> two 16-byte ds loads.
__device__ __forceinline__ v16h load_b_t(const _Float16* Wt, int ldk,
                                         int kBase, int nBase, int lane) {
  const _Float16* p =
      Wt + (nBase + (lane & 15)) * ldk + kBase + (lane >> 4) * 8;
  v8h lo = *(const v8h*)p;
  v8h hi = *(const v8h*)(p + 16);
  return cat8(lo, hi);
}

// TDM: DMA one contiguous 64KB (16384 f32) block global -> LDS.
// D# group0/group1 built per CDNA5 ISA 8.3/8.4 (count=1, data_size=4B,
// tensor_dim0 = tile_dim0 = 16384, type=2 "image"). Issued by one wave.
__device__ __forceinline__ void tdm_load_chunk(const float* gsrc,
                                               unsigned lds_off) {
  typedef unsigned int u32;
  typedef __attribute__((ext_vector_type(4))) u32 v4u_;
  typedef __attribute__((ext_vector_type(8))) int v8i_;
  typedef __attribute__((ext_vector_type(4))) int v4i_;
  unsigned long long ga = (unsigned long long)(uintptr_t)gsrc;
  v4u_ g0 = {1u,  // count=1, user descriptor, no gather
             lds_off,
             (u32)(ga & 0xffffffffu),
             (u32)((ga >> 32) & 0x01ffffffu) | (2u << 30)};  // type=2
  v8i_ g1 = {(int)0x20000,         // wg_mask=0, data_size=2 (4 bytes)
             (int)(16384u << 16),  // tensor_dim0[15:0] in bits 31:16
             (int)(1u << 16),      // tensor_dim0 hi=0 | tensor_dim1=1
             (int)(16384u << 16),  // tensor_dim1 hi=0 | tile_dim0=16384
             1,                    // tile_dim1=1, tile_dim2=0
             16384,                // tensor_dim0_stride[31:0]
             0, 0};
  v4i_ z4 = {0, 0, 0, 0};
#if defined(__clang_major__) && (__clang_major__ >= 23)
  v8i_ z8 = {0, 0, 0, 0, 0, 0, 0, 0};
  __builtin_amdgcn_tensor_load_to_lds(g0, g1, z4, z4, z8, 0);
#else
  __builtin_amdgcn_tensor_load_to_lds(g0, g1, z4, z4, 0);
#endif
}

// C/D layout: VGPR r, lane l -> row = mBase + r + 8*(l>>4), col = nBase + (l&15)
#define ST_F16      0
#define ST_F16_GELU 1
#define ST_F32_RES  2
#define ST_F32      3
#define ST_F16_T    4   // store transposed [col][ldo]: one 16B store per lane

__device__ __forceinline__ void store_tile(v8f c, int mBase, int nBase, int lane,
                                           const float* bias, _Float16* oh,
                                           float* of, int ldo, int colOff, int mode) {
  const int col = nBase + (lane & 15);
  const int r0  = mBase + 8 * (lane >> 4);
  const float bv = bias ? bias[col] : 0.0f;
  float vv[8];
#pragma unroll
  for (int r = 0; r < 8; ++r) vv[r] = c[r] + bv;

  if (mode == ST_F16_T) {
    v8h pk;
#pragma unroll
    for (int r = 0; r < 8; ++r) pk[r] = (_Float16)vv[r];
    *(v8h*)(oh + (colOff + col) * ldo + r0) = pk;  // r0 mult of 8 -> 16B aligned
    return;
  }
#pragma unroll
  for (int r = 0; r < 8; ++r) {
    int idx = (r0 + r) * ldo + colOff + col;
    if (mode == ST_F16) {
      oh[idx] = (_Float16)vv[r];
    } else if (mode == ST_F16_GELU) {
      float g = 0.5f * vv[r] * (1.0f + erff(vv[r] * 0.70710678118654752f));
      oh[idx] = (_Float16)g;
    } else if (mode == ST_F32_RES) {
      of[idx] += vv[r];
    } else {
      of[idx] = vv[r];
    }
  }
}

// Stage a global fp32 weight W[K][N] (row-major) into LDS transposed f16 [N][K].
template <int K, int N>
__device__ __forceinline__ void stage_wT(const float* __restrict__ g,
                                         _Float16* lds, int tid) {
  for (int j = tid; j < K * N; j += 256) {
    int k = j / N, n = j % N;              // coalesced global read
    lds[n * K + k] = (_Float16)g[j];
  }
}

// Generic M=256 GEMM: out[256xND] = A[256xKD] @ W[KDxND] (+bias), per mode.
// W given transposed in LDS as [ND][KD].
template <int KD, int ND>
__device__ __forceinline__ void block_gemm(const _Float16* A, int lda,
                                           const _Float16* Wt, const float* bias,
                                           _Float16* oh, float* of, int ldo,
                                           int colOff, int mode, int wave, int lane) {
#pragma unroll
  for (int mi = 0; mi < 2; ++mi) {
    const int mt = wave * 2 + mi;
    v16h a[KD / 32];
#pragma unroll
    for (int ks = 0; ks < KD / 32; ++ks)
      a[ks] = load_a_frag(A, lda, mt * 16, ks * 32, lane);
#pragma unroll
    for (int nt = 0; nt < ND / 16; ++nt) {
      v8f acc = ZERO8;
#pragma unroll
      for (int ks = 0; ks < KD / 32; ++ks) {
        v16h bf = load_b_t(Wt, KD, ks * 32, nt * 16, lane);
        acc = wmma16(a[ks], bf, acc);
      }
      store_tile(acc, mt * 16, nt * 16, lane, bias, oh, of, ldo, colOff, mode);
    }
  }
}

__global__ __launch_bounds__(256, 1) void vit_fwd(
    const float* __restrict__ x,
    const float* __restrict__ in_ln_s, const float* __restrict__ in_ln_b,
    const float* __restrict__ in_W,    const float* __restrict__ in_b,
    const float* __restrict__ in_ln2_s,const float* __restrict__ in_ln2_b,
    const float* __restrict__ msa_ln_s,const float* __restrict__ msa_ln_b,
    const float* __restrict__ Wq, const float* __restrict__ bq,
    const float* __restrict__ Wk, const float* __restrict__ bk,
    const float* __restrict__ Wv, const float* __restrict__ bv,
    const float* __restrict__ mhp_W1, const float* __restrict__ mhp_b1,
    const float* __restrict__ mhp_W2, const float* __restrict__ mhp_b2,
    const float* __restrict__ mlp_ln_s, const float* __restrict__ mlp_ln_b,
    const float* __restrict__ mlp_W1, const float* __restrict__ mlp_b1,
    const float* __restrict__ mlp_W2, const float* __restrict__ mlp_b2,
    const float* __restrict__ mlp_W3, const float* __restrict__ mlp_b3,
    const float* __restrict__ mlp_W4, const float* __restrict__ mlp_b4,
    const float* __restrict__ out_W,  const float* __restrict__ out_b,
    float* __restrict__ out) {
  extern __shared__ char smem[];
  float*    X  = (float*)(smem + OFF_X);
  _Float16* Xn = (_Float16*)(smem + OFF_XN);
  _Float16* Qh = (_Float16*)(smem + OFF_Q);
  _Float16* Kh = (_Float16*)(smem + OFF_K);
  _Float16* Vt = (_Float16*)(smem + OFF_V);   // transposed [32][256]
  _Float16* Sb = (_Float16*)(smem + OFF_S);
  _Float16* Hc = (_Float16*)(smem + OFF_H);
  _Float16* Ws = (_Float16*)(smem + OFF_W);
  float*    red = (float*)(smem + OFF_R);

  const int tid  = threadIdx.x;
  const int lane = tid & 31;
  const int wave = tid >> 5;
  const int bidx = blockIdx.x;
  const float* xb = x + (size_t)bidx * (3 * 256 * 256);

  // =================== Phase 1: patchify + LN(768) + D->L + LN(64) =========
  // Patch s=(p0,p1) holds pixels (h*16+p0, w*16+p1); feature d = c*256+h*16+w.
  // K-chunk kc needs raw pixels xb[kc*16384 .. +16384): one contiguous 64KB
  // block -> TDM it into LDS, double-buffered, 2 chunks ahead.
  if (wave == 0) {
    tdm_load_chunk(xb + 0 * 16384, OFF_STG + 0 * 65536);
    tdm_load_chunk(xb + 1 * 16384, OFF_STG + 1 * 65536);
  }

  const int p0 = tid >> 4, p1 = tid & 15;
  float sum = 0.f, sq = 0.f;
  for (int c = 0; c < 3; ++c)
    for (int hh = 0; hh < 16; ++hh) {
      const float* row = xb + ((c * 16 + hh) * 16 + p0) * 256;
#pragma unroll
      for (int w = 0; w < 16; ++w) {
        float v = row[w * 16 + p1];
        sum += v; sq += v * v;
      }
    }
  const float mean = sum * (1.f / 768.f);
  const float rstd = rsqrtf(sq * (1.f / 768.f) - mean * mean + LN_EPS);

  v8f acc[2][4];
#pragma unroll
  for (int i = 0; i < 2; ++i)
#pragma unroll
    for (int j = 0; j < 4; ++j) acc[i][j] = ZERO8;

  for (int kc = 0; kc < 12; ++kc) {  // K chunks of 64 over D=768
    if (wave == 0) {                 // chunk kc DMA complete?
      if (kc == 11) __builtin_amdgcn_s_wait_tensorcnt(0);
      else          __builtin_amdgcn_s_wait_tensorcnt(1);
    }
    __syncthreads();
    // normalized A chunk [256 x 64] from staged LDS, vectorized 16B stores
    const float* stg = (const float*)(smem + OFF_STG + (kc & 1) * 65536);
    for (int i8 = 0; i8 < 8; ++i8) {
      v8h pk;
#pragma unroll
      for (int e = 0; e < 8; ++e) {
        int i = i8 * 8 + e;
        int d = kc * 64 + i;
        float v = stg[(i >> 4) * 4096 + p0 * 256 + (i & 15) * 16 + p1];
        pk[e] = (_Float16)(((v - mean) * rstd) * in_ln_s[d] + in_ln_b[d]);
      }
      *(v8h*)(Xn + tid * 64 + i8 * 8) = pk;
    }
    // stage in_W chunk [64x64] transposed -> Ws [64N][64K]
    stage_wT<64, 64>(in_W + kc * 4096, Ws, tid);
    __syncthreads();
    // staging buffer (kc&1) is consumed: prefetch chunk kc+2 into it
    if (wave == 0 && kc + 2 < 12)
      tdm_load_chunk(xb + (kc + 2) * 16384, OFF_STG + (kc & 1) * 65536);

    v16h a[2][2];
#pragma unroll
    for (int mi = 0; mi < 2; ++mi) {
      a[mi][0] = load_a_frag(Xn, 64, (wave * 2 + mi) * 16, 0, lane);
      a[mi][1] = load_a_frag(Xn, 64, (wave * 2 + mi) * 16, 32, lane);
    }
#pragma unroll
    for (int nt = 0; nt < 4; ++nt) {
      v16h b0 = load_b_t(Ws, 64, 0, nt * 16, lane);
      v16h b1 = load_b_t(Ws, 64, 32, nt * 16, lane);
#pragma unroll
      for (int mi = 0; mi < 2; ++mi) {
        acc[mi][nt] = wmma16(a[mi][0], b0, acc[mi][nt]);
        acc[mi][nt] = wmma16(a[mi][1], b1, acc[mi][nt]);
      }
    }
    __syncthreads();
  }
#pragma unroll
  for (int mi = 0; mi < 2; ++mi)
#pragma unroll
    for (int nt = 0; nt < 4; ++nt)
      store_tile(acc[mi][nt], (wave * 2 + mi) * 16, nt * 16, lane, in_b,
                 nullptr, X, 64, 0, ST_F32);
  __syncthreads();
  {  // in_ln2 (in place) -> X = x_skip
    float4* xr4 = (float4*)(X + tid * 64);
    float m = 0.f, v = 0.f;
#pragma unroll
    for (int q = 0; q < 16; ++q) {
      float4 c = xr4[q];
      m += c.x + c.y + c.z + c.w;
    }
    m *= (1.f / 64.f);
#pragma unroll
    for (int q = 0; q < 16; ++q) {
      float4 c = xr4[q];
      v += (c.x-m)*(c.x-m) + (c.y-m)*(c.y-m) + (c.z-m)*(c.z-m) + (c.w-m)*(c.w-m);
    }
    float rs = rsqrtf(v * (1.f / 64.f) + LN_EPS);
#pragma unroll
    for (int q = 0; q < 16; ++q) {
      float4 c = xr4[q];
      int l = q * 4;
      c.x = (c.x - m) * rs * in_ln2_s[l+0] + in_ln2_b[l+0];
      c.y = (c.y - m) * rs * in_ln2_s[l+1] + in_ln2_b[l+1];
      c.z = (c.z - m) * rs * in_ln2_s[l+2] + in_ln2_b[l+2];
      c.w = (c.w - m) * rs * in_ln2_s[l+3] + in_ln2_b[l+3];
      xr4[q] = c;
    }
  }
  __syncthreads();

  // =================== Phase 2: 3 attention heads ===========================
  for (int h = 0; h < NMSA; ++h) {
    {  // per-head LN -> Xn (f16), vectorized
      const float4* xr4 = (const float4*)(X + tid * 64);
      float m = 0.f, v = 0.f;
#pragma unroll
      for (int q = 0; q < 16; ++q) {
        float4 c = xr4[q];
        m += c.x + c.y + c.z + c.w;
      }
      m *= (1.f / 64.f);
#pragma unroll
      for (int q = 0; q < 16; ++q) {
        float4 c = xr4[q];
        v += (c.x-m)*(c.x-m) + (c.y-m)*(c.y-m) + (c.z-m)*(c.z-m) + (c.w-m)*(c.w-m);
      }
      float rs = rsqrtf(v * (1.f / 64.f) + LN_EPS);
      const float* ls = msa_ln_s + h * 64;
      const float* lb = msa_ln_b + h * 64;
#pragma unroll
      for (int q = 0; q < 8; ++q) {
        v8h pk;
#pragma unroll
        for (int e = 0; e < 8; ++e) {
          int l = q * 8 + e;
          float xv = X[tid * 64 + l];
          pk[e] = (_Float16)((xv - m) * rs * ls[l] + lb[l]);
        }
        *(v8h*)(Xn + tid * 64 + q * 8) = pk;
      }
    }
    // stage Wq/Wk/Wv [64x32] transposed -> [32][64]
    stage_wT<64, 32>(Wq + h * 2048, Ws, tid);
    stage_wT<64, 32>(Wk + h * 2048, Ws + 2048, tid);
    stage_wT<64, 32>(Wv + h * 2048, Ws + 4096, tid);
    __syncthreads();
    block_gemm<64, 32>(Xn, 64, Ws,        bq + h * 32, Qh, nullptr, 32, 0, ST_F16,   wave, lane);
    block_gemm<64, 32>(Xn, 64, Ws + 2048, bk + h * 32, Kh, nullptr, 32, 0, ST_F16,   wave, lane);
    block_gemm<64, 32>(Xn, 64, Ws + 4096, bv + h * 32, Vt, nullptr, 256, 0, ST_F16_T, wave, lane);
    __syncthreads();

    for (int strip = 0; strip < 2; ++strip) {  // 128 query rows per strip
      const int mOff = strip * 128;
      {  // scores: wave owns M-tile (mOff + wave*16), all 16 N-tiles; K=E=32
        v16h aq = load_a_frag(Qh, 32, mOff + wave * 16, 0, lane);
        for (int nt = 0; nt < 16; ++nt) {
          v16h bkf = load_b_t(Kh, 32, 0, nt * 16, lane);  // B = K^T
          v8f sc = ZERO8;
          sc = wmma16(aq, bkf, sc);
          store_tile(sc, wave * 16, nt * 16, lane, nullptr, Sb, nullptr, 256, 0, ST_F16);
        }
      }
      __syncthreads();
      if (tid < 128) {  // softmax over this strip's rows (no 1/sqrt(d): faithful)
        v8h* srow = (v8h*)(Sb + tid * 256);
        float mx = -3.0e38f;
        for (int j = 0; j < 32; ++j) {
          v8h c = srow[j];
#pragma unroll
          for (int e = 0; e < 8; ++e) mx = fmaxf(mx, (float)c[e]);
        }
        float ssum = 0.f;
        for (int j = 0; j < 32; ++j) {
          v8h c = srow[j];
#pragma unroll
          for (int e = 0; e < 8; ++e) {
            float ex = __expf((float)c[e] - mx);
            ssum += ex;
            c[e] = (_Float16)ex;
          }
          srow[j] = c;
        }
        float inv = 1.f / ssum;
        for (int j = 0; j < 32; ++j) {
          v8h c = srow[j];
#pragma unroll
          for (int e = 0; e < 8; ++e) c[e] = (_Float16)((float)c[e] * inv);
          srow[j] = c;
        }
      }
      __syncthreads();
      {  // head strip = A[128x256] @ V[256x32]; B from transposed Vt
        v16h af[8];
#pragma unroll
        for (int ks = 0; ks < 8; ++ks)
          af[ks] = load_a_frag(Sb, 256, wave * 16, ks * 32, lane);
#pragma unroll
        for (int nt = 0; nt < 2; ++nt) {
          v8f a2 = ZERO8;
#pragma unroll
          for (int ks = 0; ks < 8; ++ks) {
            v16h bf = load_b_t(Vt, 256, ks * 32, nt * 16, lane);
            a2 = wmma16(af[ks], bf, a2);
          }
          store_tile(a2, mOff + wave * 16, nt * 16, lane, nullptr, Hc, nullptr,
                     96, h * 32, ST_F16);
        }
      }
      __syncthreads();
    }
  }

  // =================== Phase 3: MH projection + MLP =========================
  _Float16* T1 = Sb;
  _Float16* T2 = Sb + 16384;

  stage_wT<96, 64>(mhp_W1, Ws, tid);   // -> [64][96]
  __syncthreads();
  block_gemm<96, 64>(Hc, 96, Ws, mhp_b1, T1, nullptr, 64, 0, ST_F16, wave, lane);
  __syncthreads();
  stage_wT<64, 64>(mhp_W2, Ws, tid);
  __syncthreads();
  block_gemm<64, 64>(T1, 64, Ws, mhp_b2, nullptr, X, 64, 0, ST_F32_RES, wave, lane);
  __syncthreads();  // X = x_skip (mh-proj + residual)

  {  // MLP layernorm -> Xn
    const float4* xr4 = (const float4*)(X + tid * 64);
    float m = 0.f, v = 0.f;
#pragma unroll
    for (int q = 0; q < 16; ++q) {
      float4 c = xr4[q];
      m += c.x + c.y + c.z + c.w;
    }
    m *= (1.f / 64.f);
#pragma unroll
    for (int q = 0; q < 16; ++q) {
      float4 c = xr4[q];
      v += (c.x-m)*(c.x-m) + (c.y-m)*(c.y-m) + (c.z-m)*(c.z-m) + (c.w-m)*(c.w-m);
    }
    float rs = rsqrtf(v * (1.f / 64.f) + LN_EPS);
#pragma unroll
    for (int q = 0; q < 8; ++q) {
      v8h pk;
#pragma unroll
      for (int e = 0; e < 8; ++e) {
        int l = q * 8 + e;
        float xv = X[tid * 64 + l];
        pk[e] = (_Float16)((xv - m) * rs * mlp_ln_s[l] + mlp_ln_b[l]);
      }
      *(v8h*)(Xn + tid * 64 + q * 8) = pk;
    }
  }
  __syncthreads();

  stage_wT<64, 64>(mlp_W1, Ws, tid);
  __syncthreads();
  block_gemm<64, 64>(Xn, 64, Ws, mlp_b1, T1, nullptr, 64, 0, ST_F16, wave, lane);
  __syncthreads();
  stage_wT<64, 64>(mlp_W2, Ws, tid);
  __syncthreads();
  block_gemm<64, 64>(T1, 64, Ws, mlp_b2, T2, nullptr, 64, 0, ST_F16_GELU, wave, lane);
  __syncthreads();
  stage_wT<64, 64>(mlp_W3, Ws, tid);
  __syncthreads();
  block_gemm<64, 64>(T2, 64, Ws, mlp_b3, T1, nullptr, 64, 0, ST_F16, wave, lane);
  __syncthreads();
  stage_wT<64, 64>(mlp_W4, Ws, tid);
  __syncthreads();
  block_gemm<64, 64>(T1, 64, Ws, mlp_b4, nullptr, X, 64, 0, ST_F32_RES, wave, lane);
  __syncthreads();  // X = m + x_skip

  // =================== Phase 4: output head (16384 -> 1) ====================
  {
    const float4* xr4 = (const float4*)(X + tid * 64);
    const float4* wr4 = (const float4*)(out_W + tid * 64);
    float part = 0.f;
#pragma unroll
    for (int q = 0; q < 16; ++q) {
      float4 a = xr4[q], b = wr4[q];
      part += a.x * b.x + a.y * b.y + a.z * b.z + a.w * b.w;
    }
    red[tid] = part;
  }
  __syncthreads();
  for (int off2 = 128; off2 > 0; off2 >>= 1) {
    if (tid < off2) red[tid] += red[tid + off2];
    __syncthreads();
  }
  if (tid == 0) out[bidx] = red[0] + out_b[0];
}

extern "C" void kernel_launch(void* const* d_in, const int* in_sizes, int n_in,
                              void* d_out, int out_size, void* d_ws, size_t ws_size,
                              hipStream_t stream) {
  (void)in_sizes; (void)n_in; (void)d_ws; (void)ws_size; (void)out_size;
  const float* x        = (const float*)d_in[0];
  const float* in_ln_s  = (const float*)d_in[1];
  const float* in_ln_b  = (const float*)d_in[2];
  const float* in_W     = (const float*)d_in[3];
  const float* in_b     = (const float*)d_in[4];
  const float* in_ln2_s = (const float*)d_in[5];
  const float* in_ln2_b = (const float*)d_in[6];
  const float* msa_ln_s = (const float*)d_in[7];
  const float* msa_ln_b = (const float*)d_in[8];
  const float* Wq       = (const float*)d_in[9];
  const float* bq       = (const float*)d_in[10];
  const float* Wk       = (const float*)d_in[11];
  const float* bk       = (const float*)d_in[12];
  const float* Wv       = (const float*)d_in[13];
  const float* bv       = (const float*)d_in[14];
  const float* mhp_W1   = (const float*)d_in[15];
  const float* mhp_b1   = (const float*)d_in[16];
  const float* mhp_W2   = (const float*)d_in[17];
  const float* mhp_b2   = (const float*)d_in[18];
  const float* mlp_ln_s = (const float*)d_in[19];
  const float* mlp_ln_b = (const float*)d_in[20];
  const float* mlp_W1   = (const float*)d_in[21];
  const float* mlp_b1   = (const float*)d_in[22];
  const float* mlp_W2   = (const float*)d_in[23];
  const float* mlp_b2   = (const float*)d_in[24];
  const float* mlp_W3   = (const float*)d_in[25];
  const float* mlp_b3   = (const float*)d_in[26];
  const float* mlp_W4   = (const float*)d_in[27];
  const float* mlp_b4   = (const float*)d_in[28];
  const float* out_W    = (const float*)d_in[29];
  const float* out_b    = (const float*)d_in[30];

  hipFuncSetAttribute((const void*)vit_fwd,
                      hipFuncAttributeMaxDynamicSharedMemorySize, SMEM_BYTES);
  vit_fwd<<<256, 256, SMEM_BYTES, stream>>>(
      x, in_ln_s, in_ln_b, in_W, in_b, in_ln2_s, in_ln2_b, msa_ln_s, msa_ln_b,
      Wq, bq, Wk, bk, Wv, bv, mhp_W1, mhp_b1, mhp_W2, mhp_b2,
      mlp_ln_s, mlp_ln_b, mlp_W1, mlp_b1, mlp_W2, mlp_b2, mlp_W3, mlp_b3,
      mlp_W4, mlp_b4, out_W, out_b, (float*)d_out);
}
